// GraphConvLayer_20572893348184
// MI455X (gfx1250) — compile-verified, matrix operand order checked by reading, not run on hardware
//
#include <hip/hip_runtime.h>
#include <hip/hip_bf16.h>

// ---------------------------------------------------------------------------
// GNN message-passing layer on gfx1250 (MI455X), bf16 WMMA path.
//   msg:  [E,384] @ W1(384x128) +b1 -> relu -> @ W2(128x128) +b2 -> scatter-mean
//   upd:  [N,256] @ U1(256x128) +b1 -> relu -> @ U2(128x128) +b2
// 256-thread block (8 wave32) per 32-row tile; wave w owns N-cols [16w,16w+16)
// and computes TWO 16x16 accumulators (rows 0-15 and 16-31) that share each
// B (weight) fragment -> halves weight refetch traffic per row vs M=16 tiling.
// K-loop of v_wmma_f32_16x16x32_bf16, f32 accumulators.
// ---------------------------------------------------------------------------

typedef __attribute__((ext_vector_type(4)))  float  f32x4;
typedef __attribute__((ext_vector_type(8)))  float  v8f;
typedef __attribute__((ext_vector_type(16))) __bf16 v16bf;

union FragU { f32x4 q[2]; v16bf v; };   // 32 bytes = one 16-bit WMMA operand/lane

#define DIM   128
#define MT    32    // rows (edges / nodes) per block
#define PADA  392   // 384+8 : stride mod 64 banks = 4 dwords -> conflict-free
#define PADU  264   // 256+8
#define PADH  136   // 128+8

static __device__ __forceinline__ __bf16 f2bf(float f) {
  union { float f; unsigned u; } a; a.f = f;
  unsigned r = a.u + 0x7FFFu + ((a.u >> 16) & 1u);      // round-to-nearest-even
  union { unsigned short s; __bf16 b; } o; o.s = (unsigned short)(r >> 16);
  return o.b;
}

// ---------------------------------------------------------------------------
// Zero the scatter accumulator (d_out doubles as "summed") and the counts.
// ---------------------------------------------------------------------------
__global__ void zero_kernel(float* __restrict__ out, float* __restrict__ counts,
                            int n_out, int n_counts) {
  int stride = gridDim.x * blockDim.x;
  for (int i = blockIdx.x * blockDim.x + threadIdx.x; i < n_out;    i += stride) out[i]    = 0.f;
  for (int i = blockIdx.x * blockDim.x + threadIdx.x; i < n_counts; i += stride) counts[i] = 0.f;
}

// ---------------------------------------------------------------------------
// Pre-swizzle a (Ktiles*32 x 128) f32 weight into bf16 WMMA-B fragment order:
// tile (kt,nt) -> 512 bf16 blob; lane L holds its 16 elements contiguously
// (N = nt*16 + L%16;  K = kt*32 + (L/16)*8 + e  for e<8,  +16 for e>=8),
// matching the ISA 16-bit 32x16 B layout, so B loads are 2x16B coalesced.
// ---------------------------------------------------------------------------
__global__ void swizzle_w_kernel(const float* __restrict__ W, __bf16* __restrict__ out) {
  int b  = blockIdx.x;
  int kt = b >> 3, nt = b & 7;
  int lane  = threadIdx.x;               // 32 threads = 1 wave
  int khalf = (lane >> 4) << 3;
  int n     = nt * 16 + (lane & 15);
  __bf16* o = out + (size_t)b * 512 + lane * 16;
#pragma unroll
  for (int e = 0; e < 16; ++e) {
    int kk = (e < 8) ? (khalf + e) : (16 + khalf + (e - 8));
    o[e] = f2bf(W[(kt * 32 + kk) * DIM + n]);
  }
}

// ---------------------------------------------------------------------------
// Edge kernel: 32 edges/block. Stage [x[row] | x[col] | edge_attr] as bf16
// into LDS, run 2 WMMA GEMMs (dual 16-row tiles per wave sharing B frags),
// atomically scatter messages into summed(=d_out).
// ---------------------------------------------------------------------------
__global__ __launch_bounds__(256)
void edge_msg_kernel(const float* __restrict__ x,
                     const long long* __restrict__ ei,      // int64 [2,E]
                     const float* __restrict__ ea,
                     const __bf16* __restrict__ w1s, const float* __restrict__ b1,
                     const __bf16* __restrict__ w2s, const float* __restrict__ b2,
                     float* __restrict__ summed, float* __restrict__ counts,
                     int E) {
  __shared__ __align__(16) __bf16 sA[MT * PADA];
  __shared__ __align__(16) __bf16 sH[MT * PADH];
  __shared__ int sRow[MT], sCol[MT];

  const int t    = threadIdx.x;
  const int lane = t & 31, wave = t >> 5;
  const int e0   = blockIdx.x * MT;

  if (t < MT) {
    int e   = e0 + t;
    sRow[t] = (e < E) ? (int)ei[e]     : 0;
    sCol[t] = (e < E) ? (int)ei[E + e] : 0;
  }
  // prefetch next edge tile's edge_attr (lowers to global_prefetch_b8)
  {
    long pf = (long)(e0 + MT) * DIM + t * 16;
    if (pf < (long)E * DIM) __builtin_prefetch(&ea[pf], 0, 0);
  }
  __syncthreads();

  // stage A tile: 32 x 384 bf16 (row-major, padded rows)
  for (int i = t; i < MT * 384; i += 256) {
    int m = i / 384, c = i - m * 384;
    int e = e0 + m;
    float v = 0.f;
    if (e < E) {
      if (c < 128)      v = x[(long)sRow[m] * DIM + c];
      else if (c < 256) v = x[(long)sCol[m] * DIM + (c - 128)];
      else              v = ea[(long)e * DIM + (c - 256)];
    }
    sA[m * PADA + c] = f2bf(v);
  }
  __syncthreads();

  const int nb    = wave << 4;           // this wave's N-column base
  const int khalf = (lane >> 4) << 3;    // A/B fragment K sub-offset
  const int arow  = (lane & 15);         // A fragment row within a 16-row tile

  // ---- GEMM1: [32x384] @ W1 -> acc0/acc1, bias + relu --------------------
  v8f acc0, acc1;
  {
    float bias = b1[nb + (lane & 15)];
#pragma unroll
    for (int r = 0; r < 8; ++r) { acc0[r] = bias; acc1[r] = bias; }
  }
#pragma unroll
  for (int kt = 0; kt < 12; ++kt) {
    FragU a0, a1, b;
    const __bf16* bp = w1s + (size_t)(kt * 8 + wave) * 512 + lane * 16;
    b.q[0] = *(const f32x4*)bp;
    b.q[1] = *(const f32x4*)(bp + 8);
    const __bf16* ap0 = &sA[arow * PADA + kt * 32 + khalf];
    const __bf16* ap1 = ap0 + 16 * PADA;
    a0.q[0] = *(const f32x4*)ap0;
    a0.q[1] = *(const f32x4*)(ap0 + 16);
    a1.q[0] = *(const f32x4*)ap1;
    a1.q[1] = *(const f32x4*)(ap1 + 16);
    acc0 = __builtin_amdgcn_wmma_f32_16x16x32_bf16(false, a0.v, false, b.v,
                                                   (short)0, acc0, false, false);
    acc1 = __builtin_amdgcn_wmma_f32_16x16x32_bf16(false, a1.v, false, b.v,
                                                   (short)0, acc1, false, false);
  }
  // relu -> bf16 h tile in LDS (C/D layout: VGPR r -> M = r + 8*(lane>=16))
#pragma unroll
  for (int r = 0; r < 8; ++r) {
    int M = r + ((lane >> 4) << 3);
    sH[M * PADH + nb + (lane & 15)]             = f2bf(fmaxf(acc0[r], 0.f));
    sH[(M + 16) * PADH + nb + (lane & 15)]      = f2bf(fmaxf(acc1[r], 0.f));
  }
  __syncthreads();

  // ---- GEMM2: h[32x128] @ W2 -> messages ---------------------------------
  {
    float bias = b2[nb + (lane & 15)];
#pragma unroll
    for (int r = 0; r < 8; ++r) { acc0[r] = bias; acc1[r] = bias; }
  }
#pragma unroll
  for (int kt = 0; kt < 4; ++kt) {
    FragU a0, a1, b;
    const __bf16* bp = w2s + (size_t)(kt * 8 + wave) * 512 + lane * 16;
    b.q[0] = *(const f32x4*)bp;
    b.q[1] = *(const f32x4*)(bp + 8);
    const __bf16* ap0 = &sH[arow * PADH + kt * 32 + khalf];
    const __bf16* ap1 = ap0 + 16 * PADH;
    a0.q[0] = *(const f32x4*)ap0;
    a0.q[1] = *(const f32x4*)(ap0 + 16);
    a1.q[0] = *(const f32x4*)ap1;
    a1.q[1] = *(const f32x4*)(ap1 + 16);
    acc0 = __builtin_amdgcn_wmma_f32_16x16x32_bf16(false, a0.v, false, b.v,
                                                   (short)0, acc0, false, false);
    acc1 = __builtin_amdgcn_wmma_f32_16x16x32_bf16(false, a1.v, false, b.v,
                                                   (short)0, acc1, false, false);
  }

  // scatter-add into summed (f32 global atomics), plus edge counts
  {
    int n = nb + (lane & 15);
#pragma unroll
    for (int r = 0; r < 8; ++r) {
      int M = r + ((lane >> 4) << 3);
      int e = e0 + M;
      if (e < E)      atomicAdd(&summed[(long)sCol[M] * DIM + n],      acc0[r]);
      if (e + 16 < E) atomicAdd(&summed[(long)sCol[M + 16] * DIM + n], acc1[r]);
    }
  }
  if (t < MT) {
    int e = e0 + t;
    if (e < E) atomicAdd(&counts[sCol[t]], 1.0f);
  }
}

// ---------------------------------------------------------------------------
// Node update: 32 nodes/block. out[] holds "summed" on entry; this block is
// the only one touching its 32 rows, so read (staging) then overwrite (final).
// ---------------------------------------------------------------------------
__global__ __launch_bounds__(256)
void node_update_kernel(const float* __restrict__ x, const float* __restrict__ counts,
                        const __bf16* __restrict__ u1s, const float* __restrict__ ub1,
                        const __bf16* __restrict__ u2s, const float* __restrict__ ub2,
                        float* __restrict__ out, int N) {
  __shared__ __align__(16) __bf16 sU[MT * PADU];
  __shared__ __align__(16) __bf16 sH[MT * PADH];

  const int t    = threadIdx.x;
  const int lane = t & 31, wave = t >> 5;
  const int n0   = blockIdx.x * MT;

  // stage [x[node] | aggregated(node)] as bf16 (aggregated = summed/count)
  for (int i = t; i < MT * 256; i += 256) {
    int m = i >> 8, c = i & 255;
    int node = n0 + m;
    float v = 0.f;
    if (node < N) {
      if (c < 128) v = x[(long)node * DIM + c];
      else {
        float cnt = counts[node];
        float s   = out[(long)node * DIM + (c - 128)];
        v = (cnt > 0.f) ? s / cnt : 0.f;
      }
    }
    sU[m * PADU + c] = f2bf(v);
  }
  __syncthreads();

  const int nb    = wave << 4;
  const int khalf = (lane >> 4) << 3;
  const int arow  = (lane & 15);

  v8f acc0, acc1;
  {
    float bias = ub1[nb + (lane & 15)];
#pragma unroll
    for (int r = 0; r < 8; ++r) { acc0[r] = bias; acc1[r] = bias; }
  }
#pragma unroll
  for (int kt = 0; kt < 8; ++kt) {
    FragU a0, a1, b;
    const __bf16* bp = u1s + (size_t)(kt * 8 + wave) * 512 + lane * 16;
    b.q[0] = *(const f32x4*)bp;
    b.q[1] = *(const f32x4*)(bp + 8);
    const __bf16* ap0 = &sU[arow * PADU + kt * 32 + khalf];
    const __bf16* ap1 = ap0 + 16 * PADU;
    a0.q[0] = *(const f32x4*)ap0;
    a0.q[1] = *(const f32x4*)(ap0 + 16);
    a1.q[0] = *(const f32x4*)ap1;
    a1.q[1] = *(const f32x4*)(ap1 + 16);
    acc0 = __builtin_amdgcn_wmma_f32_16x16x32_bf16(false, a0.v, false, b.v,
                                                   (short)0, acc0, false, false);
    acc1 = __builtin_amdgcn_wmma_f32_16x16x32_bf16(false, a1.v, false, b.v,
                                                   (short)0, acc1, false, false);
  }
#pragma unroll
  for (int r = 0; r < 8; ++r) {
    int M = r + ((lane >> 4) << 3);
    sH[M * PADH + nb + (lane & 15)]        = f2bf(fmaxf(acc0[r], 0.f));
    sH[(M + 16) * PADH + nb + (lane & 15)] = f2bf(fmaxf(acc1[r], 0.f));
  }
  __syncthreads();

  {
    float bias = ub2[nb + (lane & 15)];
#pragma unroll
    for (int r = 0; r < 8; ++r) { acc0[r] = bias; acc1[r] = bias; }
  }
#pragma unroll
  for (int kt = 0; kt < 4; ++kt) {
    FragU a0, a1, b;
    const __bf16* bp = u2s + (size_t)(kt * 8 + wave) * 512 + lane * 16;
    b.q[0] = *(const f32x4*)bp;
    b.q[1] = *(const f32x4*)(bp + 8);
    const __bf16* ap0 = &sH[arow * PADH + kt * 32 + khalf];
    const __bf16* ap1 = ap0 + 16 * PADH;
    a0.q[0] = *(const f32x4*)ap0;
    a0.q[1] = *(const f32x4*)(ap0 + 16);
    a1.q[0] = *(const f32x4*)ap1;
    a1.q[1] = *(const f32x4*)(ap1 + 16);
    acc0 = __builtin_amdgcn_wmma_f32_16x16x32_bf16(false, a0.v, false, b.v,
                                                   (short)0, acc0, false, false);
    acc1 = __builtin_amdgcn_wmma_f32_16x16x32_bf16(false, a1.v, false, b.v,
                                                   (short)0, acc1, false, false);
  }

  {
    int n = nb + (lane & 15);
#pragma unroll
    for (int r = 0; r < 8; ++r) {
      int M = r + ((lane >> 4) << 3);
      int node = n0 + M;
      if (node < N)      out[(long)node * DIM + n]        = acc0[r];
      if (node + 16 < N) out[(long)(node + 16) * DIM + n] = acc1[r];
    }
  }
}

// ---------------------------------------------------------------------------
extern "C" void kernel_launch(void* const* d_in, const int* in_sizes, int n_in,
                              void* d_out, int out_size, void* d_ws, size_t ws_size,
                              hipStream_t stream) {
  if (n_in < 11) return;
  const float*     x      = (const float*)d_in[0];
  const long long* ei     = (const long long*)d_in[1];   // int64 [2,E]
  const float*     ea     = (const float*)d_in[2];
  const float*     msg_w1 = (const float*)d_in[3];
  const float*     msg_b1 = (const float*)d_in[4];
  const float*     msg_w2 = (const float*)d_in[5];
  const float*     msg_b2 = (const float*)d_in[6];
  const float*     upd_w1 = (const float*)d_in[7];
  const float*     upd_b1 = (const float*)d_in[8];
  const float*     upd_w2 = (const float*)d_in[9];
  const float*     upd_b2 = (const float*)d_in[10];
  float* out = (float*)d_out;

  const int N = in_sizes[0] / DIM;     // 50000
  const int E = in_sizes[2] / DIM;     // 640000

  // workspace layout: counts (N f32), then bf16-swizzled weights
  size_t offCounts = 0;
  size_t offW1 = ((size_t)N * 4 + 511) & ~(size_t)511;
  size_t offW2 = offW1 + (size_t)384 * 128 * 2;
  size_t offU1 = offW2 + (size_t)128 * 128 * 2;
  size_t offU2 = offU1 + (size_t)256 * 128 * 2;
  size_t need  = offU2 + (size_t)128 * 128 * 2;
  if (ws_size < need) return;

  float*  counts = (float*)((char*)d_ws + offCounts);
  __bf16* w1s    = (__bf16*)((char*)d_ws + offW1);
  __bf16* w2s    = (__bf16*)((char*)d_ws + offW2);
  __bf16* u1s    = (__bf16*)((char*)d_ws + offU1);
  __bf16* u2s    = (__bf16*)((char*)d_ws + offU2);

  zero_kernel<<<512, 256, 0, stream>>>(out, counts, N * DIM, N);

  swizzle_w_kernel<<<12 * 8, 32, 0, stream>>>(msg_w1, w1s);   // 384x128
  swizzle_w_kernel<<< 4 * 8, 32, 0, stream>>>(msg_w2, w2s);   // 128x128
  swizzle_w_kernel<<< 8 * 8, 32, 0, stream>>>(upd_w1, u1s);   // 256x128
  swizzle_w_kernel<<< 4 * 8, 32, 0, stream>>>(upd_w2, u2s);   // 128x128

  int edgeBlocks = (E + MT - 1) / MT;
  edge_msg_kernel<<<edgeBlocks, 256, 0, stream>>>(x, ei, ea, w1s, msg_b1,
                                                  w2s, msg_b2, out, counts, E);

  int nodeBlocks = (N + MT - 1) / MT;
  node_update_kernel<<<nodeBlocks, 256, 0, stream>>>(x, counts, u1s, upd_b1,
                                                     u2s, upd_b2, out, N);
}
